// alignNet_68384469287123
// MI455X (gfx1250) — matrix operand order repaired and checked by reading, not compiled, not run on hardware
//
#include <hip/hip_runtime.h>

// ---------------------------------------------------------------------------
// Types for CDNA5 WMMA (wave32, 16x16x32 bf16 -> f32)
// ---------------------------------------------------------------------------
typedef __attribute__((ext_vector_type(16))) __bf16 v16bf;
typedef __attribute__((ext_vector_type(8)))  float  v8f;

// ---------------------------------------------------------------------------
// Weight packing: fp32 [Cout][K] (K = Cin*KS*KS, row-major, matching im2col
// order k = c*KS*KS + ky*KS + kx) -> bf16 A-fragments.
// For tile (mt,kt), lane l, slot h, the value is w[mt*16 + l%16][kt*32 + ka]
// with ka = (l<16?0:16) + h   (contiguous k order, matching the B fragment).
// Grid: x covers KT*512 entries, y = mt  (no runtime division anywhere).
// ---------------------------------------------------------------------------
__global__ void pack_w_kernel(const float* __restrict__ w, __bf16* __restrict__ dst,
                              int Cout, int K, int KT)
{
    int i  = blockIdx.x * 256 + threadIdx.x;   // index within one M-tile
    int mt = blockIdx.y;
    if (i >= KT * 512) return;
    int h    = i & 15;
    int lane = (i >> 4) & 31;
    int kt   = i >> 9;
    int m  = mt * 16 + (lane & 15);
    int ka = ((lane < 16) ? 0 : 16) + h;
    int k  = kt * 32 + ka;
    float v = (m < Cout && k < K) ? w[(long)m * K + k] : 0.0f;
    dst[(long)mt * KT * 512 + i] = (__bf16)v;
}

// ---------------------------------------------------------------------------
// Implicit-GEMM conv (NCHW, pad=(KS-1)/2, square kernel KS, stride s) on WMMA.
// Wave tile: 16 output channels x 16 consecutive-x pixels; block = 4 waves.
// Grid: x = x-tile of 64, y = output row yo, z = mt*4 + b  (b = z&3, mt = z>>2)
//   -> zero integer divisions in the kernel.
// All 4 waves share the same A (weight) tile, so each 1KB A-tile is staged in
// LDS once per block (128 threads x b64), double-buffered; lanes then read
// their v16bf fragment from LDS (ds path) while the next tile streams in.
// B-fragment (32x16, KxN): lane holds pixel N = lane%16; halves h=0..15 hold
// K = kt*32 + 16*(lane/16) + h (im2col gathered on the fly, zero padded;
// consecutive lanes -> consecutive xi -> coalesced loads).
// D layout: VGPR r -> channel row M = r + 8*(lane/16), pixel N = lane%16.
// Epilogue fuses bias, optional ReLU, optional residual add.
// ---------------------------------------------------------------------------
template<int KS>
__global__ __launch_bounds__(128) void conv_wmma_kernel(
    const float* __restrict__ x, const __bf16* __restrict__ wp,
    const float* __restrict__ bias, const float* __restrict__ res,
    float* __restrict__ out,
    int Cin, int Hi, int Wi, int Cout, int Ho, int Wo,
    int stride, int relu, int KT)
{
    __shared__ __align__(16) __bf16 smA[2][512];   // double-buffered A tile (2x1KB)

    const int KSQ = KS * KS;
    const int PAD = KS >> 1;
    int tid  = threadIdx.x;
    int lane = tid & 31;
    int wv   = tid >> 5;
    int xo   = blockIdx.x * 64 + wv * 16 + (lane & 15);
    int yo   = blockIdx.y;
    int b    = blockIdx.z & 3;
    int mt   = blockIdx.z >> 2;
    bool nv  = (xo < Wo);
    int xoc  = nv ? xo : 0;
    int ybase = yo  * stride - PAD;
    int xbase = xoc * stride - PAD;

    v8f acc = {0.f, 0.f, 0.f, 0.f, 0.f, 0.f, 0.f, 0.f};
    const __bf16* wg = wp + (long)mt * KT * 512;   // this M-tile's packed weights
    const float*  xb = x + (long)b * Cin * Hi * Wi;
    int kb = (lane >> 4) * 16;

    // stage tile 0: 512 bf16 = 128 x u64
    ((unsigned long long*)&smA[0][0])[tid] = ((const unsigned long long*)wg)[tid];
    __syncthreads();

#pragma unroll 2
    for (int kt = 0; kt < KT; ++kt) {
        v16bf afrag = *(const v16bf*)&smA[kt & 1][lane * 16];
        __builtin_prefetch(wg + (long)(kt + 2) * 512, 0, 1);
        v16bf bfrag;
        int k0 = kt * 32 + kb;
#pragma unroll
        for (int h = 0; h < 16; ++h) {
            int k   = k0 + h;
            int c   = k / KSQ;            // KSQ is a compile-time constant
            int tap = k - c * KSQ;
            int ky  = tap / KS;
            int kx  = tap - ky * KS;
            float v = 0.0f;
            if (nv && c < Cin) {
                int yi = ybase + ky;
                int xi = xbase + kx;
                if (yi >= 0 && yi < Hi && xi >= 0 && xi < Wi)
                    v = xb[((long)c * Hi + yi) * Wi + xi];
            }
            bfrag[h] = (__bf16)v;
        }
        acc = __builtin_amdgcn_wmma_f32_16x16x32_bf16(
            false, afrag, false, bfrag, (short)0, acc, false, false);
        // stream next A tile into the other LDS buffer while WMMA retires
        if (kt + 1 < KT)
            ((unsigned long long*)&smA[(kt + 1) & 1][0])[tid] =
                ((const unsigned long long*)(wg + (long)(kt + 1) * 512))[tid];
        __syncthreads();
    }

    int mrow = (lane < 16) ? 0 : 8;
#pragma unroll
    for (int r = 0; r < 8; ++r) {
        int co = mt * 16 + r + mrow;
        if (nv && co < Cout) {
            float v = acc[r] + bias[co];
            if (relu) v = fmaxf(v, 0.0f);
            long oi = (((long)b * Cout + co) * Ho + yo) * Wo + xo;
            if (res) v += res[oi];
            out[oi] = v;
        }
    }
}

// ---------------------------------------------------------------------------
// Deformable sampling: thread per (b, k, y, x); loops channels, writes the
// modulated bilinear samples s[b][c*9+k][y][x] so the deform conv becomes a
// 1x1 WMMA GEMM with Cin = C*9 (K order c*9+k matches w.reshape(Cout, C*9)).
// ---------------------------------------------------------------------------
__global__ void deform_sample_kernel(const float* __restrict__ x,
                                     const float* __restrict__ off,
                                     const float* __restrict__ msk,
                                     float* __restrict__ s,
                                     int C, int H, int W)
{
    unsigned total = 4u * 9u * (unsigned)(H * W);
    unsigned i = blockIdx.x * blockDim.x + threadIdx.x;
    if (i >= total) return;
    unsigned W_ = (unsigned)W, H_ = (unsigned)H;
    int xx = (int)(i % W_);
    unsigned t = i / W_;
    int yy = (int)(t % H_); t /= H_;
    int k  = (int)(t % 9u);
    int b  = (int)(t / 9u);
    long HW  = (long)H * W;
    long pix = (long)yy * W + xx;

    float oy = off[((long)b * 18 + 2 * k    ) * HW + pix];
    float ox = off[((long)b * 18 + 2 * k + 1) * HW + pix];
    float m  = msk[((long)b * 9 + k) * HW + pix];

    float py = (float)yy + (float)(k / 3 - 1) + oy;
    float px = (float)xx + (float)(k % 3 - 1) + ox;
    float y0f = floorf(py), x0f = floorf(px);
    float wy = py - y0f, wx = px - x0f;
    int y0 = (int)y0f, x0 = (int)x0f;
    int y1 = y0 + 1,   x1 = x0 + 1;
    bool vy0 = (y0 >= 0) && (y0 < H);
    bool vy1 = (y1 >= 0) && (y1 < H);
    bool vx0 = (x0 >= 0) && (x0 < W);
    bool vx1 = (x1 >= 0) && (x1 < W);
    int y0c = min(max(y0, 0), H - 1), y1c = min(max(y1, 0), H - 1);
    int x0c = min(max(x0, 0), W - 1), x1c = min(max(x1, 0), W - 1);
    float w00 = (1.f - wy) * (1.f - wx) * ((vy0 && vx0) ? 1.f : 0.f);
    float w01 = (1.f - wy) * wx         * ((vy0 && vx1) ? 1.f : 0.f);
    float w10 = wy * (1.f - wx)         * ((vy1 && vx0) ? 1.f : 0.f);
    float w11 = wy * wx                 * ((vy1 && vx1) ? 1.f : 0.f);

    const float* xbb = x + (long)b * C * HW;
    float* sb = s + ((long)b * C * 9 + k) * HW + pix;
    long i00 = (long)y0c * W + x0c, i01 = (long)y0c * W + x1c;
    long i10 = (long)y1c * W + x0c, i11 = (long)y1c * W + x1c;
    for (int c = 0; c < C; ++c) {
        const float* xc = xbb + (long)c * HW;
        float v = w00 * xc[i00] + w01 * xc[i01] + w10 * xc[i10] + w11 * xc[i11];
        sb[(long)c * 9 * HW] = v * m;
    }
}

// ---------------------------------------------------------------------------
// Bilinear resize, align_corners=True (matches reference exactly).
// ---------------------------------------------------------------------------
__global__ void resize_ac_kernel(const float* __restrict__ x, float* __restrict__ o,
                                 int C, int H, int W, int Ho, int Wo)
{
    unsigned total = 4u * (unsigned)C * (unsigned)(Ho * Wo);
    unsigned i = blockIdx.x * blockDim.x + threadIdx.x;
    if (i >= total) return;
    unsigned Wo_ = (unsigned)Wo, Ho_ = (unsigned)Ho, C_ = (unsigned)C;
    int xo = (int)(i % Wo_);
    unsigned t = i / Wo_;
    int yo = (int)(t % Ho_); t /= Ho_;
    int cc = (int)(t % C_);
    int b  = (int)(t / C_);
    float sy = (Ho > 1) ? (float)(H - 1) / (float)(Ho - 1) : 0.f;
    float sx = (Wo > 1) ? (float)(W - 1) / (float)(Wo - 1) : 0.f;
    float ys = yo * sy, xs = xo * sx;
    int y0 = min(max((int)floorf(ys), 0), H - 2);
    int x0 = min(max((int)floorf(xs), 0), W - 2);
    float wy = ys - (float)y0, wx = xs - (float)x0;
    const float* base = x + ((long)b * C + cc) * H * W;
    float v00 = base[(long)y0 * W + x0],       v01 = base[(long)y0 * W + x0 + 1];
    float v10 = base[(long)(y0 + 1) * W + x0], v11 = base[(long)(y0 + 1) * W + x0 + 1];
    o[i] = (1.f - wy) * ((1.f - wx) * v00 + wx * v01) + wy * ((1.f - wx) * v10 + wx * v11);
}

// ---------------------------------------------------------------------------
// Channel-slice copy for concatenation.
// ---------------------------------------------------------------------------
__global__ void copy_ch_kernel(const float* __restrict__ s, float* __restrict__ d,
                               unsigned total, int C, int HW, int c0, int Cd)
{
    unsigned i = blockIdx.x * blockDim.x + threadIdx.x;
    if (i >= total) return;
    unsigned HW_ = (unsigned)HW, C_ = (unsigned)C;
    int hw = (int)(i % HW_);
    unsigned t = i / HW_;
    int cc = (int)(t % C_);
    int b  = (int)(t / C_);
    d[((long)b * Cd + c0 + cc) * HW + hw] = s[i];
}

// ===========================================================================
// Host-side orchestration
// ===========================================================================
namespace {

struct T { float* p; int C, H, W; };

struct Ctx {
    hipStream_t st;
    char*  ws;
    size_t off, cap;
    void* const* din;
    float* allocF(long n) {
        size_t a = (off + 255) & ~(size_t)255;
        off = a + (size_t)n * sizeof(float);
        return (float*)(ws + a);
    }
    void* allocB(size_t bytes) {
        size_t a = (off + 255) & ~(size_t)255;
        off = a + bytes;
        return (void*)(ws + a);
    }
    const float* P(int i) const { return (const float*)din[i]; }
};

inline long cdiv(long a, long b) { return (a + b - 1) / b; }

// Param tree flat indices (jax pytree: dict keys sorted, recursive).
enum {
    IDX_DC0B = 7, IDX_DC0W = 8, IDX_DC1B = 9, IDX_DC1W = 10, IDX_DC2B = 11, IDX_DC2W = 12,
    IDX_ENC1 = 13, IDX_ENC2 = 19, IDX_ENC3 = 25, IDX_ENC4 = 31,
    IDX_FUS0 = 37, IDX_FUS1 = 43, IDX_FUS2 = 49,
    IDX_OE0 = 55, IDX_OE1 = 75, IDX_OE2 = 95
};
// crb block: +0 b0, +1 b1, +2 b2, +3 w0, +4 w1, +5 w2
// oe  block: +0 c0b, +1 c0w, +2 c2b, +3 c2w, +4 mb, +5 mw, +6 ob, +7 ow,
//            +8 rb1(crb), +14 rb2(crb)

void conv_run(Ctx& c, T x, const float* w, const float* b, T o,
              int ks, int stride, bool relu, const float* res)
{
    int K  = x.C * ks * ks;
    int KT = (K + 31) / 32;
    int MT = (o.C + 15) / 16;
    long nfrag = (long)MT * KT * 512;
    __bf16* wp = (__bf16*)c.allocB((size_t)nfrag * sizeof(__bf16));
    pack_w_kernel<<<dim3((unsigned)cdiv((long)KT * 512, 256), (unsigned)MT), 256, 0, c.st>>>(
        w, wp, o.C, K, KT);
    dim3 grid((unsigned)cdiv(o.W, 64), (unsigned)o.H, (unsigned)(MT * 4));
    if (ks == 3)
        conv_wmma_kernel<3><<<grid, 128, 0, c.st>>>(x.p, wp, b, res, o.p,
            x.C, x.H, x.W, o.C, o.H, o.W, stride, relu ? 1 : 0, KT);
    else
        conv_wmma_kernel<1><<<grid, 128, 0, c.st>>>(x.p, wp, b, res, o.p,
            x.C, x.H, x.W, o.C, o.H, o.W, stride, relu ? 1 : 0, KT);
}

// conv3x3+ReLU (stride) then residual block: out = y + conv(relu(conv(y)))
T crb_run(Ctx& c, T x, int pb, int cout, int stride, float* outbuf)
{
    int Ho = x.H / stride, Wo = x.W / stride;
    long sz = 4L * cout * Ho * Wo;
    T y{c.allocF(sz), cout, Ho, Wo};
    conv_run(c, x, c.P(pb + 3), c.P(pb + 0), y, 3, stride, true, nullptr);
    T t{c.allocF(sz), cout, Ho, Wo};
    conv_run(c, y, c.P(pb + 4), c.P(pb + 1), t, 3, 1, true, nullptr);
    T o{outbuf ? outbuf : c.allocF(sz), cout, Ho, Wo};
    conv_run(c, t, c.P(pb + 5), c.P(pb + 2), o, 3, 1, false, y.p);
    return o;
}

void copy_run(Ctx& c, const float* s, float* d, int C, int HW, int c0, int Cd)
{
    unsigned tot = 4u * (unsigned)C * (unsigned)HW;
    copy_ch_kernel<<<dim3((tot + 255) / 256), 256, 0, c.st>>>(s, d, tot, C, HW, c0, Cd);
}

void resize_run(Ctx& c, const float* x, float* o, int C, int H, int W, int Ho, int Wo)
{
    unsigned tot = 4u * (unsigned)C * (unsigned)(Ho * Wo);
    resize_ac_kernel<<<dim3((tot + 255) / 256), 256, 0, c.st>>>(x, o, C, H, W, Ho, Wo);
}

void offset_est_run(Ctx& c, T f, int pb, const float* prevp, T& off, T& m, int nf)
{
    int H = f.H, W = f.W;
    T x0{c.allocF(4L * nf * H * W), nf, H, W};
    conv_run(c, f, c.P(pb + 1), c.P(pb + 0), x0, 1, 1, true, nullptr);   // 1x1 + relu
    T r1 = crb_run(c, x0, pb + 8, nf, 1, nullptr);
    T r2 = crb_run(c, r1, pb + 14, nf, 1, nullptr);
    T xin = r2;
    if (prevp) {
        int na = nf + 18;
        T cc{c.allocF(4L * na * H * W), na, H, W};
        copy_run(c, r2.p, cc.p, nf, H * W, 0, na);
        copy_run(c, prevp, cc.p, 18, H * W, nf, na);
        xin = cc;
    }
    T x2{c.allocF(4L * nf * H * W), nf, H, W};
    conv_run(c, xin, c.P(pb + 3), c.P(pb + 2), x2, 3, 1, true, nullptr); // c2 + relu
    off = T{c.allocF(4L * 18 * H * W), 18, H, W};
    conv_run(c, x2, c.P(pb + 7), c.P(pb + 6), off, 3, 1, false, nullptr); // offsets
    m = T{c.allocF(4L * 9 * H * W), 9, H, W};
    conv_run(c, x2, c.P(pb + 5), c.P(pb + 4), m, 3, 1, false, nullptr);   // mask (raw)
}

T deform_run(Ctx& c, T x, T off, T m, const float* w, const float* b,
             int cout, float* outbuf)
{
    int C = x.C, H = x.H, W = x.W;
    T s{c.allocF(4L * C * 9 * H * W), C * 9, H, W};
    unsigned tot = 4u * 9u * (unsigned)(H * W);
    deform_sample_kernel<<<dim3((tot + 255) / 256), 256, 0, c.st>>>(
        x.p, off.p, m.p, s.p, C, H, W);
    T o{outbuf ? outbuf : c.allocF(4L * cout * H * W), cout, H, W};
    conv_run(c, s, w, b, o, 1, 1, false, nullptr); // 1x1 over C*9 channels
    return o;
}

} // namespace

extern "C" void kernel_launch(void* const* d_in, const int* in_sizes, int n_in,
                              void* d_out, int out_size, void* d_ws, size_t ws_size,
                              hipStream_t stream)
{
    (void)in_sizes; (void)n_in; (void)out_size;
    Ctx c{stream, (char*)d_ws, 0, ws_size, d_in};

    const float* F0_l2 = (const float*)d_in[0];
    const float* F0_l4 = (const float*)d_in[1];
    const float* F0_l8 = (const float*)d_in[2];
    const float* F1_l2 = (const float*)d_in[3];
    const float* F1_l4 = (const float*)d_in[4];
    const float* F1_l8 = (const float*)d_in[5];
    float* out = (float*)d_out;

    const long F02sz = 4L * 32 * 192 * 192;
    const long F01sz = 4L * 64 * 96 * 96;
    float* outF02 = out;
    float* outF01 = out + F02sz;
    float* outF00 = out + F02sz + F01sz;

    // Persistent buffers (live across stages)
    float* x12p   = c.allocF(4L * 32 * 192 * 192);
    float* x14p   = c.allocF(4L * 64 * 96 * 96);
    float* x18p   = c.allocF(4L * 128 * 48 * 48);
    float* F00p   = c.allocF(4L * 64 * 48 * 48);
    float* off0up = c.allocF(4L * 18 * 96 * 96);
    float* off1up = c.allocF(4L * 18 * 192 * 192);
    size_t pm = c.off;

    // ---- Event encoder pyramid ----
    {
        T e0{(float*)d_in[6], 16, 384, 384};
        T x = crb_run(c, e0, IDX_ENC1, 32, 1, nullptr);
        crb_run(c, x, IDX_ENC2, 32, 2, x12p);
        T x12{x12p, 32, 192, 192};
        crb_run(c, x12, IDX_ENC3, 64, 2, x14p);
        T x14{x14p, 64, 96, 96};
        crb_run(c, x14, IDX_ENC4, 128, 2, x18p);
    }
    c.off = pm;

    // ---- Level 0 (1/8) ----
    {
        T f{c.allocF(4L * 384 * 48 * 48), 384, 48, 48};
        copy_run(c, F0_l8, f.p, 128, 48 * 48, 0, 384);
        copy_run(c, x18p,  f.p, 128, 48 * 48, 128, 384);
        copy_run(c, F1_l8, f.p, 128, 48 * 48, 256, 384);
        T off0, mm0;
        offset_est_run(c, f, IDX_OE0, nullptr, off0, mm0, 384);
        T F0l8{(float*)d_in[2], 128, 48, 48};
        deform_run(c, F0l8, off0, mm0, c.P(IDX_DC0W), c.P(IDX_DC0B), 64, F00p);
        resize_run(c, off0.p, off0up, 18, 48, 48, 96, 96);
        T F00{F00p, 64, 48, 48};
        crb_run(c, F00, IDX_FUS0, 128, 1, outF00);
    }
    c.off = pm;

    // ---- Level 1 (1/4) ----
    {
        T f{c.allocF(4L * 192 * 96 * 96), 192, 96, 96};
        copy_run(c, F0_l4, f.p, 64, 96 * 96, 0, 192);
        copy_run(c, x14p,  f.p, 64, 96 * 96, 64, 192);
        copy_run(c, F1_l4, f.p, 64, 96 * 96, 128, 192);
        T off1, mm1;
        offset_est_run(c, f, IDX_OE1, off0up, off1, mm1, 192);
        T F0l4{(float*)d_in[1], 64, 96, 96};
        T F01d = deform_run(c, F0l4, off1, mm1, c.P(IDX_DC1W), c.P(IDX_DC1B), 32, nullptr);
        resize_run(c, off1.p, off1up, 18, 96, 96, 192, 192);
        float* F00up = c.allocF(4L * 64 * 96 * 96);
        resize_run(c, F00p, F00up, 64, 48, 48, 96, 96);
        T cc{c.allocF(4L * 96 * 96 * 96), 96, 96, 96};
        copy_run(c, F01d.p, cc.p, 32, 96 * 96, 0, 96);
        copy_run(c, F00up,  cc.p, 64, 96 * 96, 32, 96);
        crb_run(c, cc, IDX_FUS1, 64, 1, outF01);
    }
    c.off = pm;

    // ---- Level 2 (1/2) ----
    {
        T f{c.allocF(4L * 96 * 192 * 192), 96, 192, 192};
        copy_run(c, F0_l2, f.p, 32, 192 * 192, 0, 96);
        copy_run(c, x12p,  f.p, 32, 192 * 192, 32, 96);
        copy_run(c, F1_l2, f.p, 32, 192 * 192, 64, 96);
        T off2, mm2;
        offset_est_run(c, f, IDX_OE2, off1up, off2, mm2, 96);
        T F0l2{(float*)d_in[0], 32, 192, 192};
        T F02d = deform_run(c, F0l2, off2, mm2, c.P(IDX_DC2W), c.P(IDX_DC2B), 32, nullptr);
        float* F01up = c.allocF(4L * 64 * 192 * 192);
        resize_run(c, outF01, F01up, 64, 96, 96, 192, 192);
        T cc{c.allocF(4L * 96 * 192 * 192), 96, 192, 192};
        copy_run(c, F02d.p, cc.p, 32, 192 * 192, 0, 96);
        copy_run(c, F01up,  cc.p, 64, 192 * 192, 32, 96);
        crb_run(c, cc, IDX_FUS2, 32, 1, outF02);
    }
}
